// TinyGPT_37237366456640
// MI455X (gfx1250) — matrix operand order repaired and checked by reading, not compiled, not run on hardware
//
#include <hip/hip_runtime.h>
#include <hip/hip_bf16.h>
#include <hip/hip_fp16.h>
#include <stdint.h>

// TinyGPT forward on gfx1250 (MI455X).
//  - All GEMMs + attention matmuls: v_wmma_f32_16x16x32_f16 (wave32, f32 acc)
//  - GEMM tiles staged LDS via Tensor Data Mover (tensor_load_to_lds +
//    s_wait_tensorcnt), double buffered, BK=64.
//  - Attention K-tiles staged via global_load_async_to_lds_b128 + s_wait_asynccnt.
//  - Weights pre-converted fp32 -> f16 and pre-transposed to [N][K] once per call.
//
// Shapes: V=50257, L=6, H=8, D=512, HD=64, B=4, T=1024 -> M = B*T = 4096.

#define TT    1024
#define DD    512
#define LL    6
#define HH    8
#define HDIM  64
#define BB    4
#define MROWS (BB * TT)   // 4096
#define VV    50257
#define VPAD  50304       // 393 * 128

typedef __attribute__((ext_vector_type(16))) _Float16     v16h;
typedef __attribute__((ext_vector_type(8)))  float        v8f;
typedef __attribute__((ext_vector_type(4)))  unsigned int u32x4;
typedef __attribute__((ext_vector_type(8)))  unsigned int u32x8;

union Frag {
    v16h  v;
    u32x4 q[2];
};

__device__ inline v8f wmma_f16(const Frag& a, const Frag& b, v8f c) {
    return __builtin_amdgcn_wmma_f32_16x16x32_f16(
        false, a.v, false, b.v, (short)0, c, false, false);
}

// ---------------------------------------------------------------------------
// CDNA5 async data movement (inline asm; see CDNA5 ISA)
// ---------------------------------------------------------------------------

// Per-lane async 16B copy global -> LDS (tracked by ASYNCcnt).
__device__ inline void async_ld16(void* lds, const void* g) {
    asm volatile("global_load_async_to_lds_b128 %0, %1, off"
                 :: "v"((unsigned)(uintptr_t)lds),
                    "v"((unsigned long long)(uintptr_t)g)
                 : "memory");
}
__device__ inline void wait_async0() {
    asm volatile("s_wait_asynccnt 0x0" ::: "memory");
}

// TDM 2D tile load: tile_d0 (contiguous, elems) x tile_d1 rows of f16,
// row stride in elements; LDS gets 16B padding after every 128B (matches the
// 72-half padded LDS rows used by the WMMA fragment readers).
__device__ inline void tdm_load_2d_f16(unsigned lds_addr, const _Float16* gtile,
                                       unsigned tile_d0, unsigned tile_d1,
                                       unsigned long long stride_elems,
                                       unsigned tensor_d0, unsigned tensor_d1) {
    unsigned long long ga = (unsigned long long)(uintptr_t)gtile;
    u32x4 g0;
    g0[0] = 1u;                                           // count=1 (valid D#)
    g0[1] = lds_addr;                                     // lds byte address
    g0[2] = (unsigned)(ga & 0xffffffffu);                 // global_addr lo
    g0[3] = (unsigned)((ga >> 32) & 0x1ffffffu) | (2u << 30); // addr hi | type=2
    u32x8 g1;
    g1[0] = (1u << 16)      // data_size = 2 bytes
          | (1u << 20)      // pad_enable
          | (4u << 22)      // pad_interval: 32 DWORDs (128B)
          | (3u << 25);     // pad_amount:   4 DWORDs (16B)
    g1[1] = (tensor_d0 & 0xffffu) << 16;                  // tensor_dim0 lo16
    g1[2] = ((tensor_d0 >> 16) & 0xffffu) | ((tensor_d1 & 0xffffu) << 16);
    g1[3] = ((tensor_d1 >> 16) & 0xffffu) | (tile_d0 << 16);
    g1[4] = tile_d1 & 0xffffu;                            // tile_dim1 (dim2=0)
    g1[5] = (unsigned)(stride_elems & 0xffffffffu);       // dim0_stride lo32
    g1[6] = (unsigned)((stride_elems >> 32) & 0xffffu);   // hi16 (dim1_stride=0)
    g1[7] = 0u;
    asm volatile("tensor_load_to_lds %0, %1" :: "s"(g0), "s"(g1) : "memory");
}

// ---------------------------------------------------------------------------
// Weight convert + transpose: W[K][N] f32 -> Wt[Npad][K] f16 (zero-padded rows)
// ---------------------------------------------------------------------------
__global__ void __launch_bounds__(256)
k_wconv(const float* __restrict__ W, _Float16* __restrict__ Wt, int K, int N) {
    __shared__ float tile[32][33];
    int nb = blockIdx.x * 32;
    int kb = blockIdx.y * 32;
    int tx = threadIdx.x & 31;
    int ty = threadIdx.x >> 5;          // 0..7
#pragma unroll
    for (int i = 0; i < 4; i++) {
        int k = kb + ty + i * 8;
        int n = nb + tx;
        tile[ty + i * 8][tx] = (n < N) ? W[(long)k * N + n] : 0.f;
    }
    __syncthreads();
#pragma unroll
    for (int i = 0; i < 4; i++) {
        int n = nb + ty + i * 8;
        Wt[(long)n * K + kb + tx] = (_Float16)tile[tx][ty + i * 8];
    }
}

// ---------------------------------------------------------------------------
// Embedding
// ---------------------------------------------------------------------------
__global__ void __launch_bounds__(256)
k_embed(const int* __restrict__ idx, const float* __restrict__ tok,
        const float* __restrict__ pos, float* __restrict__ x) {
    int row = blockIdx.x;
    int t   = row & (TT - 1);
    int id  = idx[row];
    int c   = threadIdx.x * 2;
    const float* te = tok + (long)id * DD;
    const float* pe = pos + (long)t * DD;
    float*       xo = x   + (long)row * DD;
    xo[c]     = te[c]     + pe[c];
    xo[c + 1] = te[c + 1] + pe[c + 1];
}

// ---------------------------------------------------------------------------
// LayerNorm: one wave32 per row; f16 output (A operand of the next GEMM).
// ---------------------------------------------------------------------------
__global__ void __launch_bounds__(256)
k_ln(const float* __restrict__ x, const float* __restrict__ g,
     const float* __restrict__ b, _Float16* __restrict__ out) {
    int lane = threadIdx.x & 31;
    int wave = threadIdx.x >> 5;
    int row  = blockIdx.x * 8 + wave;
    const float* xr = x + (long)row * DD;

    float v[16];
    float s = 0.f;
#pragma unroll
    for (int i = 0; i < 16; i++) { v[i] = xr[lane + i * 32]; s += v[i]; }
#pragma unroll
    for (int m = 1; m < 32; m <<= 1) s += __shfl_xor(s, m, 32);
    float mean = s * (1.0f / DD);

    float vs = 0.f;
#pragma unroll
    for (int i = 0; i < 16; i++) { float d = v[i] - mean; vs += d * d; }
#pragma unroll
    for (int m = 1; m < 32; m <<= 1) vs += __shfl_xor(vs, m, 32);
    float rstd = rsqrtf(vs * (1.0f / DD) + 1e-5f);

    _Float16* orow = out + (long)row * DD;
#pragma unroll
    for (int i = 0; i < 16; i++) {
        int c = lane + i * 32;
        orow[c] = (_Float16)((v[i] - mean) * rstd * g[c] + b[c]);
    }
}

// ---------------------------------------------------------------------------
// WMMA GEMM: C[M x N] = A_f16[M x K] * Bt_f16[N x K]^T  (+ epilogue)
//   Block tile 128x128, 8 waves (2x4), wave tile 64x32, BK=64 (2 wmma k-steps).
//   Both tiles staged by TDM into double-buffered padded LDS (row = 72 halves).
// Epilogues: 0 = f16 bias (QKV) | 1 = f32 bias+resid | 2 = f16 gelu(bias) |
//            3 = f32 plain (LM head)
// ---------------------------------------------------------------------------
template <int EPI>
__global__ void __launch_bounds__(256)
k_gemm(const _Float16* __restrict__ A, const _Float16* __restrict__ Bt,
       const float* bias, float* resid, void* outv,
       int M, int N, int K) {
    __shared__ _Float16 As[2][128][72];
    __shared__ _Float16 Bs[2][128][72];

    int tid   = threadIdx.x;
    int lane  = tid & 31;
    int wave  = tid >> 5;
    int waveM = wave >> 2;
    int waveN = wave & 3;
    int hi    = lane >> 4;
    int l15   = lane & 15;
    long blockN = (long)blockIdx.x * 128;
    long blockM = (long)blockIdx.y * 128;

    v8f acc[4][2];
#pragma unroll
    for (int mt = 0; mt < 4; mt++)
#pragma unroll
        for (int nt = 0; nt < 2; nt++)
#pragma unroll
            for (int r = 0; r < 8; r++) acc[mt][nt][r] = 0.f;

    int ksteps = K >> 6;

    auto stage = [&](int buf, int k0) {
        if (wave == 0) {
            tdm_load_2d_f16((unsigned)(uintptr_t)&As[buf][0][0],
                            A + blockM * K + k0, 64, 128,
                            (unsigned long long)K, (unsigned)K, 1u << 30);
            tdm_load_2d_f16((unsigned)(uintptr_t)&Bs[buf][0][0],
                            Bt + blockN * K + k0, 64, 128,
                            (unsigned long long)K, (unsigned)K, 1u << 30);
        }
    };

    stage(0, 0);
    for (int kt = 0; kt < ksteps; kt++) {
        int cur = kt & 1;
        bool more = (kt + 1 < ksteps);
        if (more) stage(cur ^ 1, (kt + 1) << 6);
        if (wave == 0) {
            if (more) __builtin_amdgcn_s_wait_tensorcnt(2);
            else      __builtin_amdgcn_s_wait_tensorcnt(0);
        }
        __syncthreads();
#pragma unroll
        for (int ks = 0; ks < 2; ks++) {
            Frag af[4], bf[2];
#pragma unroll
            for (int mt = 0; mt < 4; mt++) {
                const _Float16* arow = &As[cur][waveM * 64 + mt * 16 + l15][ks * 32];
                af[mt].q[0] = *(const u32x4*)(arow + hi * 8);
                af[mt].q[1] = *(const u32x4*)(arow + hi * 8 + 16);
            }
#pragma unroll
            for (int nt = 0; nt < 2; nt++) {
                const _Float16* brow = &Bs[cur][waveN * 32 + nt * 16 + l15][ks * 32];
                bf[nt].q[0] = *(const u32x4*)(brow + hi * 8);
                bf[nt].q[1] = *(const u32x4*)(brow + hi * 8 + 16);
            }
#pragma unroll
            for (int mt = 0; mt < 4; mt++)
#pragma unroll
                for (int nt = 0; nt < 2; nt++)
                    acc[mt][nt] = wmma_f16(af[mt], bf[nt], acc[mt][nt]);
        }
        __syncthreads();
    }

    // Epilogue. C/D layout: lane l15 = col, rows r + 8*hi.
#pragma unroll
    for (int mt = 0; mt < 4; mt++) {
#pragma unroll
        for (int nt = 0; nt < 2; nt++) {
            long col = blockN + waveN * 32 + nt * 16 + l15;
            if (col < N) {
                float bv = (EPI == 3) ? 0.f : bias[col];
#pragma unroll
                for (int r = 0; r < 8; r++) {
                    long row = blockM + waveM * 64 + mt * 16 + r + hi * 8;
                    float val = acc[mt][nt][r] + bv;
                    if (EPI == 0) {
                        ((_Float16*)outv)[row * N + col] = (_Float16)val;
                    } else if (EPI == 1) {
                        ((float*)outv)[row * N + col] = val + resid[row * N + col];
                    } else if (EPI == 2) {
                        float o = 0.5f * val * (1.f + erff(val * 0.70710678f));
                        ((_Float16*)outv)[row * N + col] = (_Float16)o;
                    } else {
                        ((float*)outv)[row * N + col] = val;
                    }
                }
            }
        }
    }
}

// ---------------------------------------------------------------------------
// Flash attention. qkv row = b*T + t, col = which*512 + h*64 + d (f16).
// Block = (b, h, 128-query chunk); 8 waves x one 16-query tile. Keys in
// blocks of 32: K staged async-to-LDS, V transposed manually, QK^T and P*V
// via WMMA, online softmax (row stats replicated across 16-lane groups).
// ---------------------------------------------------------------------------
__global__ void __launch_bounds__(256)
k_attn(const _Float16* __restrict__ qkv, _Float16* __restrict__ o16) {
    __shared__ _Float16 Ks[32][72];
    __shared__ _Float16 Vt[64][40];
    __shared__ _Float16 Ps[8][16][40];

    int tid  = threadIdx.x;
    int lane = tid & 31;
    int wave = tid >> 5;
    int hi   = lane >> 4;
    int l15  = lane & 15;
    int bh   = blockIdx.x;
    int b    = bh >> 3;
    int h    = bh & 7;
    int qBase = blockIdx.y * 128;
    int qtile = qBase + wave * 16;
    long rowBase = (long)b * TT;

    const _Float16* Qg = qkv + rowBase * (3 * DD) + h * HDIM;
    const _Float16* Kg = Qg + DD;
    const _Float16* Vg = Qg + 2 * DD;

    Frag aQ[2];
    {
        const _Float16* qrow = Qg + (long)(qtile + l15) * (3 * DD);
#pragma unroll
        for (int ks = 0; ks < 2; ks++) {
            aQ[ks].q[0] = *(const u32x4*)(qrow + ks * 32 + hi * 8);
            aQ[ks].q[1] = *(const u32x4*)(qrow + ks * 32 + hi * 8 + 16);
        }
    }

    v8f accO[4];
#pragma unroll
    for (int d = 0; d < 4; d++)
#pragma unroll
        for (int r = 0; r < 8; r++) accO[d][r] = 0.f;
    float mrow[8], lrow[8];
#pragma unroll
    for (int r = 0; r < 8; r++) { mrow[r] = -1e30f; lrow[r] = 0.f; }

    int numKB = (qBase + 128) >> 5;
    int fk = tid >> 3;                 // key 0..31
    int fd = (tid & 7) << 3;           // 8 dims

    for (int kb = 0; kb < numKB; kb++) {
        int kbBase = kb << 5;
        __syncthreads();
        { // K: async 16B chunks; V: manual transpose
            async_ld16(&Ks[fk][fd], Kg + (long)(kbBase + fk) * (3 * DD) + fd);
            const _Float16* vp = Vg + (long)(kbBase + fk) * (3 * DD) + fd;
#pragma unroll
            for (int i = 0; i < 8; i++) Vt[fd + i][fk] = vp[i];
        }
        wait_async0();
        __syncthreads();

        bool active = (kbBase <= qtile + 15);   // wave-uniform
        if (active) {
            v8f s[2];
#pragma unroll
            for (int t = 0; t < 2; t++) {
#pragma unroll
                for (int r = 0; r < 8; r++) s[t][r] = 0.f;
                const _Float16* krow = &Ks[t * 16 + l15][0];
                Frag bk0, bk1;
                bk0.q[0] = *(const u32x4*)(krow + hi * 8);
                bk0.q[1] = *(const u32x4*)(krow + hi * 8 + 16);
                bk1.q[0] = *(const u32x4*)(krow + 32 + hi * 8);
                bk1.q[1] = *(const u32x4*)(krow + 32 + hi * 8 + 16);
                s[t] = wmma_f16(aQ[0], bk0, s[t]);
                s[t] = wmma_f16(aQ[1], bk1, s[t]);
            }
#pragma unroll
            for (int r = 0; r < 8; r++) {
                int query = qtile + r + hi * 8;
                float s0 = s[0][r] * 0.125f;
                float s1 = s[1][r] * 0.125f;
                bool m0 = (kbBase + l15)      > query;
                bool m1 = (kbBase + 16 + l15) > query;
                if (m0) s0 = -1e30f;
                if (m1) s1 = -1e30f;
                float mm = fmaxf(s0, s1);
#pragma unroll
                for (int mk = 1; mk < 16; mk <<= 1) mm = fmaxf(mm, __shfl_xor(mm, mk, 16));
                float mnew  = fmaxf(mrow[r], mm);
                float alpha = __expf(mrow[r] - mnew);
                float p0 = m0 ? 0.f : __expf(s0 - mnew);
                float p1 = m1 ? 0.f : __expf(s1 - mnew);
                float rs = p0 + p1;
#pragma unroll
                for (int mk = 1; mk < 16; mk <<= 1) rs += __shfl_xor(rs, mk, 16);
                lrow[r] = lrow[r] * alpha + rs;
                mrow[r] = mnew;
#pragma unroll
                for (int d = 0; d < 4; d++) accO[d][r] *= alpha;
                Ps[wave][r + hi * 8][l15]      = (_Float16)p0;
                Ps[wave][r + hi * 8][16 + l15] = (_Float16)p1;
            }
        }
        __syncthreads();
        if (active) {
            Frag aP;
            const _Float16* prow = &Ps[wave][l15][0];
            aP.q[0] = *(const u32x4*)(prow + hi * 8);
            aP.q[1] = *(const u32x4*)(prow + hi * 8 + 16);
#pragma unroll
            for (int d = 0; d < 4; d++) {
                Frag bv;
                const _Float16* vrow = &Vt[d * 16 + l15][0];
                bv.q[0] = *(const u32x4*)(vrow + hi * 8);
                bv.q[1] = *(const u32x4*)(vrow + hi * 8 + 16);
                accO[d] = wmma_f16(aP, bv, accO[d]);
            }
        }
    }

#pragma unroll
    for (int d = 0; d < 4; d++) {
#pragma unroll
        for (int r = 0; r < 8; r++) {
            long row = rowBase + qtile + r + hi * 8;
            int  col = h * HDIM + d * 16 + l15;
            o16[row * DD + col] = (_Float16)(accO[d][r] / lrow[r]);
        }
    }
}

// ---------------------------------------------------------------------------
// Host driver
// ---------------------------------------------------------------------------
extern "C" void kernel_launch(void* const* d_in, const int* in_sizes, int n_in,
                              void* d_out, int out_size, void* d_ws, size_t ws_size,
                              hipStream_t stream) {
    const int*   idx    = (const int*)  d_in[0];
    const float* tok    = (const float*)d_in[1];
    const float* pos    = (const float*)d_in[2];
    const float* ln1_g  = (const float*)d_in[3];
    const float* ln1_b  = (const float*)d_in[4];
    const float* qkv_w  = (const float*)d_in[5];
    const float* qkv_b  = (const float*)d_in[6];
    const float* proj_w = (const float*)d_in[7];
    const float* proj_b = (const float*)d_in[8];
    const float* ln2_g  = (const float*)d_in[9];
    const float* ln2_b  = (const float*)d_in[10];
    const float* ff1_w  = (const float*)d_in[11];
    const float* ff1_b  = (const float*)d_in[12];
    const float* ff2_w  = (const float*)d_in[13];
    const float* ff2_b  = (const float*)d_in[14];
    const float* lnf_g  = (const float*)d_in[15];
    const float* lnf_b  = (const float*)d_in[16];
    const float* head_w = (const float*)d_in[17];

    // workspace carve (~136 MB)
    char* ws = (char*)d_ws;
    float*    x     = (float*)ws;     ws += (size_t)MROWS * DD * 4;
    _Float16* h16   = (_Float16*)ws;  ws += (size_t)MROWS * DD * 2;
    _Float16* qkv16 = (_Float16*)ws;  ws += (size_t)MROWS * 3 * DD * 2;
    _Float16* o16   = (_Float16*)ws;  ws += (size_t)MROWS * DD * 2;
    _Float16* ff16  = (_Float16*)ws;  ws += (size_t)MROWS * 4 * DD * 2;
    _Float16* qkvT  = (_Float16*)ws;  ws += (size_t)LL * 3 * DD * DD * 2;
    _Float16* projT = (_Float16*)ws;  ws += (size_t)LL * DD * DD * 2;
    _Float16* ff1T  = (_Float16*)ws;  ws += (size_t)LL * 4 * DD * DD * 2;
    _Float16* ff2T  = (_Float16*)ws;  ws += (size_t)LL * 4 * DD * DD * 2;
    _Float16* headT = (_Float16*)ws;  ws += (size_t)VPAD * DD * 2;

    // ---- one-pass weight convert + transpose (fp32 [K][N] -> f16 [Npad][K])
    for (int l = 0; l < LL; l++) {
        k_wconv<<<dim3(48, 16), 256, 0, stream>>>(
            qkv_w + (size_t)l * DD * 3 * DD, qkvT + (size_t)l * 3 * DD * DD, DD, 3 * DD);
        k_wconv<<<dim3(16, 16), 256, 0, stream>>>(
            proj_w + (size_t)l * DD * DD, projT + (size_t)l * DD * DD, DD, DD);
        k_wconv<<<dim3(64, 16), 256, 0, stream>>>(
            ff1_w + (size_t)l * DD * 4 * DD, ff1T + (size_t)l * 4 * DD * DD, DD, 4 * DD);
        k_wconv<<<dim3(16, 64), 256, 0, stream>>>(
            ff2_w + (size_t)l * 4 * DD * DD, ff2T + (size_t)l * 4 * DD * DD, 4 * DD, DD);
    }
    k_wconv<<<dim3(VPAD / 32, 16), 256, 0, stream>>>(head_w, headT, DD, VV);

    k_embed<<<MROWS, 256, 0, stream>>>(idx, tok, pos, x);

    for (int l = 0; l < LL; l++) {
        k_ln<<<MROWS / 8, 256, 0, stream>>>(x, ln1_g + l * DD, ln1_b + l * DD, h16);
        k_gemm<0><<<dim3(12, 32), 256, 0, stream>>>(
            h16, qkvT + (size_t)l * 3 * DD * DD, qkv_b + l * 3 * DD,
            nullptr, qkv16, MROWS, 3 * DD, DD);
        k_attn<<<dim3(BB * HH, TT / 128), 256, 0, stream>>>(qkv16, o16);
        k_gemm<1><<<dim3(4, 32), 256, 0, stream>>>(
            o16, projT + (size_t)l * DD * DD, proj_b + l * DD,
            x, x, MROWS, DD, DD);
        k_ln<<<MROWS / 8, 256, 0, stream>>>(x, ln2_g + l * DD, ln2_b + l * DD, h16);
        k_gemm<2><<<dim3(16, 32), 256, 0, stream>>>(
            h16, ff1T + (size_t)l * 4 * DD * DD, ff1_b + l * 4 * DD,
            nullptr, ff16, MROWS, 4 * DD, DD);
        k_gemm<1><<<dim3(4, 32), 256, 0, stream>>>(
            ff16, ff2T + (size_t)l * 4 * DD * DD, ff2_b + l * DD,
            x, x, MROWS, DD, 4 * DD);
    }

    k_ln<<<MROWS / 8, 256, 0, stream>>>(x, lnf_g, lnf_b, h16);
    k_gemm<3><<<dim3(VPAD / 128, 32), 256, 0, stream>>>(
        h16, headT, nullptr, nullptr, d_out, MROWS, VV, DD);
}
